// SelfAttention_79577154060955
// MI455X (gfx1250) — compile-verified, matrix operand order checked by reading, not run on hardware
//
#include <hip/hip_runtime.h>

typedef _Float16 v16h __attribute__((ext_vector_type(16)));
typedef _Float16 v8h  __attribute__((ext_vector_type(8)));
typedef float    v8f  __attribute__((ext_vector_type(8)));
typedef float    v4f  __attribute__((ext_vector_type(4)));

#define DDIM 1024
#define SDIM 2048
#define BATCH 4
#define LDS_PITCH 40   // 32 data halves + 8 pad halves (80B rows -> bank-friendly)

// ---- addrspace(1) access helpers: force global_load/global_store (not flat) ----
__device__ __forceinline__ v8h load_g_v8h(const _Float16* p) {
    return *(const __attribute__((address_space(1))) v8h*)(unsigned long long)p;
}
__device__ __forceinline__ v4f load_g_v4f(const float* p) {
    return *(const __attribute__((address_space(1))) v4f*)(unsigned long long)p;
}
__device__ __forceinline__ _Float16 load_g_f16(const _Float16* p) {
    return *(const __attribute__((address_space(1))) _Float16*)(unsigned long long)p;
}
__device__ __forceinline__ void store_g_f16(_Float16* p, _Float16 v) {
    *(__attribute__((address_space(1))) _Float16*)(unsigned long long)p = v;
}
__device__ __forceinline__ void store_g_f32(float* p, float v) {
    *(__attribute__((address_space(1))) float*)(unsigned long long)p = v;
}
__device__ __forceinline__ void store_g_v8h(_Float16* p, v8h v) {
    *(__attribute__((address_space(1))) v8h*)(unsigned long long)p = v;
}

__device__ __forceinline__ v16h cat8(v8h lo, v8h hi) {
    v16h r;
#pragma unroll
    for (int i = 0; i < 8; ++i) { r[i] = lo[i]; r[i + 8] = hi[i]; }
    return r;
}

// Block-level GEMM: 256 threads (8 waves as 2x4), block tile 128(M) x 256(N),
// C = A(MxK, row-major) * B(NxK, row-major)^T, K = nsteps*32.
// A/B tiles are staged through double-buffered LDS; each wave computes a
// 64x64 tile with 16 v_wmma_f32_16x16x32_f16 per K-step.
__device__ __forceinline__ void block_gemm_128x256(
    const _Float16* __restrict__ A, const _Float16* __restrict__ B,
    int lda, int ldb, int nsteps, int blockM, int blockN, v8f acc[4][4])
{
    __shared__ _Float16 As[2][128 * LDS_PITCH];   // 2 x 10 KB
    __shared__ _Float16 Bs[2][256 * LDS_PITCH];   // 2 x 20 KB

    const int t     = threadIdx.x;
    const int w     = t >> 5;
    const int waveM = w >> 2;         // 0..1
    const int waveN = w & 3;          // 0..3
    const int lane  = t & 31;
    const int half  = lane >> 4;      // K-subgroup selector per WMMA layout
    const int r     = lane & 15;

    // --- global staging addresses (coalesced-ish, 32B/64B per thread/row) ---
    const int arow  = t >> 1;              // 0..127
    const int acol  = (t & 1) * 16;        // element offset within 32-elt row
    const _Float16* pA = A + (size_t)(blockM + arow) * lda + acol;
    const _Float16* pB = B + (size_t)(blockN + t) * ldb;   // 64B per thread row

    v8h a_reg[2], b_reg[4];

    // prologue: fetch tile 0 and stage to LDS buffer 0
    a_reg[0] = load_g_v8h(pA);
    a_reg[1] = load_g_v8h(pA + 8);
#pragma unroll
    for (int c = 0; c < 4; ++c) b_reg[c] = load_g_v8h(pB + c * 8);
    *(v8h*)&As[0][arow * LDS_PITCH + acol]     = a_reg[0];
    *(v8h*)&As[0][arow * LDS_PITCH + acol + 8] = a_reg[1];
#pragma unroll
    for (int c = 0; c < 4; ++c)
        *(v8h*)&Bs[0][t * LDS_PITCH + c * 8] = b_reg[c];
    __syncthreads();

    for (int ks = 0; ks < nsteps; ++ks) {
        const int  cur  = ks & 1;
        const bool more = (ks + 1) < nsteps;

        // issue next tile's global loads early: latency hides under WMMAs
        if (more) {
            const _Float16* qa = pA + (ks + 1) * 32;
            a_reg[0] = load_g_v8h(qa);
            a_reg[1] = load_g_v8h(qa + 8);
            const _Float16* qb = pB + (ks + 1) * 32;
#pragma unroll
            for (int c = 0; c < 4; ++c) b_reg[c] = load_g_v8h(qb + c * 8);
        }

        // read WMMA fragments from LDS (layout matches 16x16x32 f16 operands)
        v16h afrag[4], bfrag[4];
#pragma unroll
        for (int i = 0; i < 4; ++i) {
            const _Float16* s =
                &As[cur][(waveM * 64 + 16 * i + r) * LDS_PITCH + half * 8];
            afrag[i] = cat8(*(const v8h*)s, *(const v8h*)(s + 16));
        }
#pragma unroll
        for (int j = 0; j < 4; ++j) {
            const _Float16* s =
                &Bs[cur][(waveN * 64 + 16 * j + r) * LDS_PITCH + half * 8];
            bfrag[j] = cat8(*(const v8h*)s, *(const v8h*)(s + 16));
        }

#pragma unroll
        for (int i = 0; i < 4; ++i)
#pragma unroll
            for (int j = 0; j < 4; ++j)
                acc[i][j] = __builtin_amdgcn_wmma_f32_16x16x32_f16(
                    false, afrag[i], false, bfrag[j], (short)0, acc[i][j],
                    false, false);

        // stage prefetched tile into the other LDS buffer
        if (more) {
            const int nxt = cur ^ 1;
            *(v8h*)&As[nxt][arow * LDS_PITCH + acol]     = a_reg[0];
            *(v8h*)&As[nxt][arow * LDS_PITCH + acol + 8] = a_reg[1];
#pragma unroll
            for (int c = 0; c < 4; ++c)
                *(v8h*)&Bs[nxt][t * LDS_PITCH + c * 8] = b_reg[c];
        }
        __syncthreads();
    }
}

// Vectorized fp32 -> fp16 conversion (8 elements / thread). n % 8 == 0.
__global__ void cvt_f32_f16_kernel(const float* __restrict__ in,
                                   _Float16* __restrict__ out, int n)
{
    int i = (blockIdx.x * 256 + threadIdx.x) * 8;
    if (i + 8 <= n) {
        v4f f0 = load_g_v4f(in + i);
        v4f f1 = load_g_v4f(in + i + 4);
        v8h h;
#pragma unroll
        for (int k = 0; k < 4; ++k) {
            h[k]     = (_Float16)f0[k];
            h[k + 4] = (_Float16)f1[k];
        }
        store_g_v8h(out + i, h);
    }
}

// GEMM1: qkv = x16[8192,1024] @ W16[3072,1024]^T.
// Q is pre-scaled by 1/sqrt(D)=1/32; V is stored transposed per batch.
__global__ __launch_bounds__(256) void qkv_gemm_kernel(
    const _Float16* __restrict__ x16, const _Float16* __restrict__ w16,
    _Float16* __restrict__ Qb, _Float16* __restrict__ Kb,
    _Float16* __restrict__ Vt)
{
    const int w  = threadIdx.x >> 5;
    const int m0 = blockIdx.y * 128 + (w >> 2) * 64;
    const int n0 = blockIdx.x * 256 + (w & 3) * 64;
    v8f acc[4][4] = {};
    block_gemm_128x256(x16, w16, DDIM, DDIM, DDIM / 32,
                       blockIdx.y * 128, blockIdx.x * 256, acc);

    const int lane = threadIdx.x & 31;
    const int col  = lane & 15;
    const int roff = (lane >> 4) * 8;
#pragma unroll
    for (int i = 0; i < 4; ++i)
#pragma unroll
        for (int j = 0; j < 4; ++j) {
            const int n = n0 + 16 * j + col;
#pragma unroll
            for (int v = 0; v < 8; ++v) {
                const int m = m0 + 16 * i + roff + v;
                float val = acc[i][j][v];
                if (n < DDIM) {
                    store_g_f16(&Qb[(size_t)m * DDIM + n], (_Float16)(val * 0.03125f));
                } else if (n < 2 * DDIM) {
                    store_g_f16(&Kb[(size_t)m * DDIM + (n - DDIM)], (_Float16)val);
                } else {
                    const int b = m >> 11, s = m & (SDIM - 1);
                    store_g_f16(&Vt[(size_t)b * DDIM * SDIM +
                                    (size_t)(n - 2 * DDIM) * SDIM + s],
                                (_Float16)val);
                }
            }
        }
}

// GEMM2: logits = Q[2048,1024] @ K[2048,1024]^T per batch (scale baked into Q).
__global__ __launch_bounds__(256) void logits_gemm_kernel(
    const _Float16* __restrict__ Qb, const _Float16* __restrict__ Kb,
    _Float16* __restrict__ S16)
{
    const int b = blockIdx.z;
    const _Float16* A = Qb + (size_t)b * SDIM * DDIM;
    const _Float16* B = Kb + (size_t)b * SDIM * DDIM;
    _Float16* C = S16 + (size_t)b * SDIM * SDIM;

    const int w  = threadIdx.x >> 5;
    const int m0 = blockIdx.y * 128 + (w >> 2) * 64;
    const int n0 = blockIdx.x * 256 + (w & 3) * 64;
    v8f acc[4][4] = {};
    block_gemm_128x256(A, B, DDIM, DDIM, DDIM / 32,
                       blockIdx.y * 128, blockIdx.x * 256, acc);

    const int lane = threadIdx.x & 31;
    const int col  = lane & 15;
    const int roff = (lane >> 4) * 8;
#pragma unroll
    for (int i = 0; i < 4; ++i)
#pragma unroll
        for (int j = 0; j < 4; ++j) {
            const int n = n0 + 16 * j + col;
#pragma unroll
            for (int v = 0; v < 8; ++v) {
                const int m = m0 + 16 * i + roff + v;
                store_g_f16(&C[(size_t)m * SDIM + n], (_Float16)acc[i][j][v]);
            }
        }
}

// Row softmax, in place on f16 logits (fp32 math). One block per row.
__global__ __launch_bounds__(256) void softmax_kernel(_Float16* __restrict__ S16)
{
    const int row = blockIdx.x;                 // 0 .. BATCH*SDIM-1
    _Float16* p = S16 + (size_t)row * SDIM;
    const int tid = threadIdx.x;

    float v[8];
    float mx = -3.0e38f;
#pragma unroll
    for (int i = 0; i < 8; ++i) {
        v[i] = (float)load_g_f16(&p[tid + i * 256]);
        mx = fmaxf(mx, v[i]);
    }
    __shared__ float red[256];
    red[tid] = mx;
    __syncthreads();
    for (int s = 128; s > 0; s >>= 1) {
        if (tid < s) red[tid] = fmaxf(red[tid], red[tid + s]);
        __syncthreads();
    }
    mx = red[0];
    __syncthreads();

    float sum = 0.f;
#pragma unroll
    for (int i = 0; i < 8; ++i) {
        v[i] = __expf(v[i] - mx);
        sum += v[i];
    }
    red[tid] = sum;
    __syncthreads();
    for (int s = 128; s > 0; s >>= 1) {
        if (tid < s) red[tid] += red[tid + s];
        __syncthreads();
    }
    const float inv = 1.0f / red[0];
#pragma unroll
    for (int i = 0; i < 8; ++i)
        store_g_f16(&p[tid + i * 256], (_Float16)(v[i] * inv));
}

// GEMM3: y = P[2048,2048] @ Vt[1024,2048]^T per batch, fp32 output.
__global__ __launch_bounds__(256) void av_gemm_kernel(
    const _Float16* __restrict__ P, const _Float16* __restrict__ Vt,
    float* __restrict__ out)
{
    const int b = blockIdx.z;
    const _Float16* A = P + (size_t)b * SDIM * SDIM;
    const _Float16* B = Vt + (size_t)b * DDIM * SDIM;
    float* C = out + (size_t)b * SDIM * DDIM;

    const int w  = threadIdx.x >> 5;
    const int m0 = blockIdx.y * 128 + (w >> 2) * 64;
    const int n0 = blockIdx.x * 256 + (w & 3) * 64;
    v8f acc[4][4] = {};
    block_gemm_128x256(A, B, SDIM, SDIM, SDIM / 32,
                       blockIdx.y * 128, blockIdx.x * 256, acc);

    const int lane = threadIdx.x & 31;
    const int col  = lane & 15;
    const int roff = (lane >> 4) * 8;
#pragma unroll
    for (int i = 0; i < 4; ++i)
#pragma unroll
        for (int j = 0; j < 4; ++j) {
            const int n = n0 + 16 * j + col;
#pragma unroll
            for (int v = 0; v < 8; ++v) {
                const int m = m0 + 16 * i + roff + v;
                store_g_f32(&C[(size_t)m * DDIM + n], acc[i][j][v]);
            }
        }
}

extern "C" void kernel_launch(void* const* d_in, const int* in_sizes, int n_in,
                              void* d_out, int out_size, void* d_ws, size_t ws_size,
                              hipStream_t stream)
{
    const float* x = (const float*)d_in[0];   // [4,2048,1024] f32
    const float* W = (const float*)d_in[1];   // [3072,1024]  f32
    float* out = (float*)d_out;               // [4,2048,1024] f32

    const size_t M = (size_t)BATCH * SDIM;    // 8192 rows
    char* ws = (char*)d_ws;
    size_t off = 0;
    _Float16* x16 = (_Float16*)(ws + off); off += M * DDIM * 2;                  // 16 MB
    _Float16* w16 = (_Float16*)(ws + off); off += (size_t)3 * DDIM * DDIM * 2;   //  6 MB
    _Float16* Qb  = (_Float16*)(ws + off); off += M * DDIM * 2;                  // 16 MB
    _Float16* Kb  = (_Float16*)(ws + off); off += M * DDIM * 2;                  // 16 MB
    _Float16* Vt  = (_Float16*)(ws + off); off += M * DDIM * 2;                  // 16 MB
    _Float16* S16 = (_Float16*)(ws + off); off += (size_t)BATCH * SDIM * SDIM * 2; // 33 MB
    (void)ws_size; (void)in_sizes; (void)n_in; (void)out_size;

    // 1) fp32 -> fp16 conversion of x and W (each reused many times downstream)
    {
        int nx = (int)(M * DDIM);
        int nw = 3 * DDIM * DDIM;
        cvt_f32_f16_kernel<<<(nx / 8 + 255) / 256, 256, 0, stream>>>(x, x16, nx);
        cvt_f32_f16_kernel<<<(nw / 8 + 255) / 256, 256, 0, stream>>>(W, w16, nw);
    }

    // 2) QKV projection: [8192,1024] x [3072,1024]^T ; block tile 128x256
    qkv_gemm_kernel<<<dim3(3 * DDIM / 256, M / 128), 256, 0, stream>>>(x16, w16, Qb, Kb, Vt);

    // 3) logits = Q K^T per batch: 2048x2048, K=1024
    logits_gemm_kernel<<<dim3(SDIM / 256, SDIM / 128, BATCH), 256, 0, stream>>>(Qb, Kb, S16);

    // 4) softmax in place, one block per row
    softmax_kernel<<<dim3((unsigned)(BATCH * SDIM)), 256, 0, stream>>>(S16);

    // 5) y = P V per batch: 2048x1024, K=2048
    av_gemm_kernel<<<dim3(DDIM / 256, SDIM / 128, BATCH), 256, 0, stream>>>(S16, Vt, out);
}
